// LE_HCL_new_40827959116391
// MI455X (gfx1250) — matrix operand order repaired and checked by reading, not compiled, not run on hardware
//
#include <hip/hip_runtime.h>
#include <stdint.h>

// ---------------------------------------------------------------------------
// Types
// ---------------------------------------------------------------------------
typedef __attribute__((ext_vector_type(16))) __bf16   v16bf;
typedef __attribute__((ext_vector_type(8)))  float    v8f;
typedef __attribute__((ext_vector_type(4)))  unsigned u32x4;
typedef __attribute__((ext_vector_type(2)))  unsigned u32x2;
typedef __attribute__((ext_vector_type(8)))  int      i32x8;
typedef __attribute__((ext_vector_type(4)))  int      i32x4;
typedef __attribute__((ext_vector_type(2)))  float    f32x2;
typedef __attribute__((ext_vector_type(4)))  float    f32x4;

#define DEVINL __device__ __forceinline__

union FragAB { u32x4 q[2]; v16bf v; };

DEVINL unsigned f2bf1(float f) {               // fp32 -> bf16 (RNE), as u16
  unsigned u = __builtin_bit_cast(unsigned, f);
  return (u + 0x7FFFu + ((u >> 16) & 1u)) >> 16;
}
DEVINL unsigned pack_bf(float lo, float hi) {  // two fp32 -> packed bf16x2
  return f2bf1(lo) | (f2bf1(hi) << 16);
}
DEVINL float bf2f(unsigned short h) {
  unsigned u = ((unsigned)h) << 16;
  return __builtin_bit_cast(float, u);
}

DEVINL v8f wmma_bf16(const FragAB& a, const FragAB& b, v8f c) {
  return __builtin_amdgcn_wmma_f32_16x16x32_bf16(
      false, a.v, false, b.v, (short)0, c, false, false);
}

// ---------------------------------------------------------------------------
// TDM: 2D bf16 tile (global -> LDS) with LDS row padding.
//   D# per CDNA5 ISA. data_size=2B.
//   pad_interval code c -> 2^(c+1) dwords; pad_amount code a -> a+1 dwords.
// Issued by one wave; completion via TENSORcnt + workgroup barrier.
// Toolchain here is the 6-arg builtin (clang-23 / therock headers).
// ---------------------------------------------------------------------------
DEVINL void tdm_load_2d_bf16(unsigned lds_byte_addr, const void* gaddr,
                             unsigned tensor_d0, unsigned tensor_d1,
                             unsigned tile_d0, unsigned tile_d1,
                             unsigned d0_stride,
                             unsigned pad_icode, unsigned pad_acode)
{
  unsigned long long ga = (unsigned long long)gaddr;
  u32x4 g0;
  g0[0] = 1u;                                   // count=1 (valid descriptor)
  g0[1] = lds_byte_addr;                        // lds_addr
  g0[2] = (unsigned)(ga & 0xFFFFFFFFu);         // global_addr[31:0]
  g0[3] = (unsigned)((ga >> 32) & 0x01FFFFFFu)  // global_addr[56:32]
        | (2u << 30);                           // type=2 ("image")
  i32x8 g1;
  g1[0] = (int)((1u << 16)                      // data_size = 2 bytes
              | (1u << 20)                      // pad_enable
              | (pad_icode << 22)               // pad_interval
              | (pad_acode << 25));             // pad_amount
  g1[1] = (int)((tensor_d0 & 0xFFFFu) << 16);   // tensor_dim0[15:0]
  g1[2] = (int)((tensor_d0 >> 16) | ((tensor_d1 & 0xFFFFu) << 16));
  g1[3] = (int)((tensor_d1 >> 16) | (tile_d0 << 16));   // tile_dim0
  g1[4] = (int)(tile_d1 & 0xFFFFu);             // tile_dim1 (tile_dim2=0 -> 2D)
  g1[5] = (int)d0_stride;                       // tensor_dim0_stride[31:0]
  g1[6] = 0;
  g1[7] = 0;
  i32x4 gz4 = {0, 0, 0, 0};
  i32x8 gz8 = {0, 0, 0, 0, 0, 0, 0, 0};
  __builtin_amdgcn_tensor_load_to_lds(g0, g1, gz4, gz4, gz8, 0);
}

// Problem constants
constexpr int  BATCH = 2;
constexpr int  CC    = 192;          // channels
constexpr int  HEADS = 6;
constexpr int  NPIX  = 256 * 256;    // 65536
constexpr int  MQKV  = 3 * CC;       // 576

// ---------------------------------------------------------------------------
// GEMM (WMMA bf16): Out[M, 65536] = A[M,192] x B[192,65536], per batch (z).
// Block tile 64(M) x 128(N); 8 waves, each 16x64; K loop 6 x 32.
// A row pad 100 dwords, B row pad 68 dwords (strides ≡ 4 mod 32 dwords ->
// conflict-free half-wave b128 reads). B panel loaded by TDM when bf16.
// ---------------------------------------------------------------------------
template<bool A_F32, bool B_F32, bool OUT_F32, bool B_TDM>
__global__ __launch_bounds__(256)
void gemm192_wmma(const void* __restrict__ Ag, const void* __restrict__ Bg,
                  void* __restrict__ Og,
                  size_t aStride, size_t bStride, size_t oStride)
{
  constexpr int AP = 100;     // dwords/row (96 used = 192 bf16)
  constexpr int BP = 68;      // dwords/row (64 used = 128 bf16)
  constexpr int BOFF = 64 * AP;                 // dword offset of B panel
  __shared__ unsigned smem[64 * AP + 192 * BP]; // single array -> LDS offset 0
  unsigned* a_lds = smem;
  unsigned* b_lds = smem + BOFF;

  const int t       = threadIdx.x;
  const int rowBase = blockIdx.y * 64;
  const int colBase = blockIdx.x * 128;
  const char* Ab = (const char*)Ag + (size_t)blockIdx.z * aStride;
  const char* Bb = (const char*)Bg + (size_t)blockIdx.z * bStride;
  char*       Ob = (char*)Og       + (size_t)blockIdx.z * oStride;
  const int lane = t & 31, w = t >> 5;

  // ---- B panel via TDM (bf16 source): 192 rows x 128 bf16, pad 64dw+4dw ----
  if (B_TDM && w == 0) {
    tdm_load_2d_bf16(BOFF * 4u,
                     (const unsigned short*)Bb + colBase,
                     /*tensor_d0=*/NPIX, /*tensor_d1=*/192,
                     /*tile_d0=*/128, /*tile_d1=*/192,
                     /*d0_stride=*/NPIX,
                     /*pad_icode(64dw)=*/5, /*pad_acode(4dw)=*/3);
  }

  // ---- stage A: 64 rows x 192 (96 dwords/row), vectorized ----
  if (A_F32) {
    const float* A = (const float*)Ab;
    for (int idx = t; idx < 64 * 48; idx += 256) {      // 48 f32x4 per row
      const int r = idx / 48, c4 = idx % 48;
      f32x4 f = *(const f32x4*)(A + (long)(rowBase + r) * 192 + c4 * 4);
      u32x2 p = {pack_bf(f.x, f.y), pack_bf(f.z, f.w)};
      *(u32x2*)&a_lds[r * AP + c4 * 2] = p;
    }
  } else {
    const unsigned* A = (const unsigned*)Ab;
    for (int idx = t; idx < 64 * 24; idx += 256) {      // 24 u32x4 per row
      const int r = idx / 24, c4 = idx % 24;
      u32x4 v = *(const u32x4*)(A + (long)(rowBase + r) * 96 + c4 * 4);
      *(u32x4*)&a_lds[r * AP + c4 * 4] = v;
    }
  }
  // ---- stage B manually when f32 source (fused fp32->bf16 convert) ----
  if (B_F32) {
    const float* B = (const float*)Bb;
    for (int idx = t; idx < 192 * 32; idx += 256) {     // 32 f32x4 per row
      const int r = idx >> 5, c4 = idx & 31;
      f32x4 f = *(const f32x4*)(B + (long)r * NPIX + colBase + c4 * 4);
      u32x2 p = {pack_bf(f.x, f.y), pack_bf(f.z, f.w)};
      *(u32x2*)&b_lds[r * BP + c4 * 2] = p;
    }
  }
  if (B_TDM && w == 0) __builtin_amdgcn_s_wait_tensorcnt(0);
  __syncthreads();

  const int msub = w & 3;          // 4 waves along M
  const int nsub = w >> 2;         // 2 waves along N
  v8f zero = {0.f,0.f,0.f,0.f,0.f,0.f,0.f,0.f};
  v8f acc[4] = {zero, zero, zero, zero};

  const int arow = msub * 16 + (lane & 15);
  const int asel = (lane >> 4) << 2;          // 0 / +4 dwords (K+8)

  #pragma unroll
  for (int kk = 0; kk < 6; ++kk) {
    const int k0 = kk * 32;
    FragAB af;
    af.q[0] = *(const u32x4*)&a_lds[arow * AP + (k0 >> 1) + asel];
    af.q[1] = *(const u32x4*)&a_lds[arow * AP + (k0 >> 1) + asel + 8];
    const unsigned short* bl = (const unsigned short*)b_lds;
    #pragma unroll
    for (int s = 0; s < 4; ++s) {
      FragAB bf;
      const int bn = nsub * 64 + s * 16 + (lane & 15);
      #pragma unroll
      for (int e = 0; e < 16; ++e) {
        const int kq = k0 + ((e & 8) << 1) + ((lane >> 4) << 3) + (e & 7);
        bf.v[e] = __builtin_bit_cast(__bf16, bl[kq * (BP * 2) + bn]);
      }
      acc[s] = wmma_bf16(af, bf, acc[s]);
    }
  }

  // C/D layout: VGPR r -> M=r (lanes 0..15), M=r+8 (lanes 16..31); N=lane%16
  #pragma unroll
  for (int s = 0; s < 4; ++s) {
    #pragma unroll
    for (int r = 0; r < 8; ++r) {
      const long m = rowBase + msub * 16 + r + ((lane >> 4) << 3);
      const long n = colBase + nsub * 64 + s * 16 + (lane & 15);
      const float val = acc[s][r];
      if (OUT_F32) ((float*)Ob)[m * NPIX + n] = val;
      else ((unsigned short*)Ob)[m * NPIX + n] = (unsigned short)f2bf1(val);
    }
  }
}

// ---------------------------------------------------------------------------
// Depthwise 3x3 (SAME, zero pad) on qkv (bf16 in, bf16 out).
//   q (ch<192)  -> qkvd normal layout [b][ch][n]
//   k (192..383)-> transposed kT: [b][head][n][32]  (for Gram B-fragments)
//   v (384..575)-> qkvd normal layout
// Also emits per-(b,ch,tile) sum-of-squares partials for q/k (l2norm).
// ---------------------------------------------------------------------------
__global__ __launch_bounds__(256)
void dwconv_kernel(const unsigned short* __restrict__ qkv,
                   const float* __restrict__ dww,         // [576,9] f32
                   unsigned short* __restrict__ outbuf,   // q/kT/v bf16
                   float* __restrict__ sumsqP)            // [b][384][32]
{
  const int x    = threadIdx.x;          // column 0..255
  const int tile = blockIdx.x;           // 0..31
  const int ch   = blockIdx.y;           // 0..575
  const int b    = blockIdx.z;
  const int y0   = tile * 8;
  const unsigned short* in = qkv + ((long)(b * MQKV + ch) << 16);

  float wgt[9];
  #pragma unroll
  for (int i = 0; i < 9; ++i) wgt[i] = dww[ch * 9 + i];

  float r0[3], r1[3], r2[3];
  auto loadRow = [&](int y, float* r) {
    if (y < 0 || y > 255) { r[0] = r[1] = r[2] = 0.f; return; }
    const unsigned short* row = in + ((long)y << 8);
    r[0] = (x > 0)   ? bf2f(row[x - 1]) : 0.f;
    r[1] = bf2f(row[x]);
    r[2] = (x < 255) ? bf2f(row[x + 1]) : 0.f;
  };
  loadRow(y0 - 1, r0);
  loadRow(y0,     r1);

  float ssq = 0.f;
  #pragma unroll
  for (int yy = 0; yy < 8; ++yy) {
    const int y = y0 + yy;
    loadRow(y + 1, r2);
    const float o = wgt[0]*r0[0] + wgt[1]*r0[1] + wgt[2]*r0[2]
                  + wgt[3]*r1[0] + wgt[4]*r1[1] + wgt[5]*r1[2]
                  + wgt[6]*r2[0] + wgt[7]*r2[1] + wgt[8]*r2[2];
    const long n = ((long)y << 8) | x;
    long idx;
    if (ch < 192) {
      idx = ((long)(b * MQKV + ch) << 16) + n;
    } else if (ch < 384) {
      const int hd = (ch - 192) >> 5, dd = (ch - 192) & 31;
      idx = ((long)(b * MQKV + 192) << 16) + ((long)hd << 21) + (n << 5) + dd;
    } else {
      idx = ((long)(b * MQKV + ch) << 16) + n;
    }
    outbuf[idx] = (unsigned short)f2bf1(o);
    if (ch < 384) ssq += o * o;
    #pragma unroll
    for (int i = 0; i < 3; ++i) { r0[i] = r1[i]; r1[i] = r2[i]; }
  }

  if (ch < 384) {
    __shared__ float red[256];
    red[x] = ssq;
    __syncthreads();
    #pragma unroll
    for (int s = 128; s > 0; s >>= 1) {
      if (x < s) red[x] += red[x + s];
      __syncthreads();
    }
    if (x == 0) sumsqP[(((long)b * 384 + ch) << 5) + tile] = red[0];
  }
}

// ---------------------------------------------------------------------------
// Gram kernel (WMMA bf16): per (b,head,slab) partial  attn += q_slab @ kT_slab
// q: [32 c][N] bf16 (A frags), kT: [N][32 d] bf16 (B frags, lane = d column).
// 2048 n per block, staged 512 at a time via two TDM tile loads.
// Deterministic: per-slab partials, per-wave LDS tiles for cross-wave reduce.
// ---------------------------------------------------------------------------
__global__ __launch_bounds__(256)
void gram_kernel(const unsigned short* __restrict__ qkvd,
                 float* __restrict__ gramP)   // [b][head][slab(32)][1024]
{
  constexpr int QP = 260;     // dwords/row q (256 used = 512 bf16)
  constexpr int KP = 20;      // dwords/row kT (16 used = 32 bf16)
  constexpr int QOFF = 0;
  constexpr int KOFF = 32 * QP;            // 8320 dwords
  constexpr int ROFF = KOFF + 512 * KP;    // 18560 dwords
  __shared__ unsigned smem[ROFF + 8 * 1024];
  unsigned* q_lds = smem + QOFF;
  unsigned* k_lds = smem + KOFF;
  float*    red   = (float*)(smem + ROFF);   // [8][1024]

  const int t    = threadIdx.x;
  const int slab = blockIdx.x;   // 0..31 (2048 n each)
  const int hd   = blockIdx.y;
  const int b    = blockIdx.z;
  const unsigned short* qbase = qkvd + ((long)(b * MQKV + hd * 32) << 16);
  const unsigned short* kbase = qkvd + ((long)(b * MQKV + 192) << 16) + ((long)hd << 21);
  const int lane = t & 31, w = t >> 5;

  v8f zero = {0.f,0.f,0.f,0.f,0.f,0.f,0.f,0.f};
  v8f acc[2][2] = {{zero, zero}, {zero, zero}};

  for (int s0 = slab * 2048; s0 < slab * 2048 + 2048; s0 += 512) {
    if (w == 0) {
      // q tile: 32 rows x 512 bf16 (256 dw + 4 dw pad -> QP)
      tdm_load_2d_bf16(QOFF * 4u, qbase + s0,
                       /*tensor_d0=*/NPIX, /*tensor_d1=*/32,
                       /*tile_d0=*/512, /*tile_d1=*/32,
                       /*d0_stride=*/NPIX,
                       /*pad_icode(256dw)=*/7, /*pad_acode(4dw)=*/3);
      // kT tile: 512 rows x 32 bf16 (16 dw + 4 dw pad -> KP)
      tdm_load_2d_bf16(KOFF * 4u, kbase + (long)s0 * 32,
                       /*tensor_d0=*/32, /*tensor_d1=*/NPIX,
                       /*tile_d0=*/32, /*tile_d1=*/512,
                       /*d0_stride=*/32,
                       /*pad_icode(16dw)=*/3, /*pad_acode(4dw)=*/3);
      __builtin_amdgcn_s_wait_tensorcnt(0);
    }
    __syncthreads();

    #pragma unroll
    for (int c = 0; c < 2; ++c) {                       // 2 K-chunks per wave
      const int n0 = (w * 2 + c) * 32;
      FragAB aq[2], bk[2];
      #pragma unroll
      for (int ch_ = 0; ch_ < 2; ++ch_) {
        const int ar  = ch_ * 16 + (lane & 15);
        const int off = (n0 >> 1) + ((lane >> 4) << 2);
        aq[ch_].q[0] = *(const u32x4*)&q_lds[ar * QP + off];
        aq[ch_].q[1] = *(const u32x4*)&q_lds[ar * QP + off + 8];
      }
      const unsigned short* kl = (const unsigned short*)k_lds;
      #pragma unroll
      for (int j = 0; j < 2; ++j)
        #pragma unroll
        for (int e = 0; e < 16; ++e) {
          const int kq = n0 + ((e & 8) << 1) + ((lane >> 4) << 3) + (e & 7);
          bk[j].v[e] = __builtin_bit_cast(__bf16, kl[kq * (KP * 2) + j * 16 + (lane & 15)]);
        }
      #pragma unroll
      for (int i = 0; i < 2; ++i)
        #pragma unroll
        for (int j = 0; j < 2; ++j)
          acc[i][j] = wmma_bf16(aq[i], bk[j], acc[i][j]);
    }
    __syncthreads();
  }

  // deterministic cross-wave reduce via per-wave LDS tiles
  #pragma unroll
  for (int i = 0; i < 2; ++i)
    #pragma unroll
    for (int j = 0; j < 2; ++j)
      #pragma unroll
      for (int r = 0; r < 8; ++r) {
        const int cc = i * 16 + r + ((lane >> 4) << 3);
        const int dd = j * 16 + (lane & 15);
        red[w * 1024 + cc * 32 + dd] = acc[i][j][r];
      }
  __syncthreads();
  float* dst = gramP + ((((long)(b * HEADS + hd)) << 5) + slab) * 1024;
  for (int d = t; d < 1024; d += 256) {
    float s = 0.f;
    #pragma unroll
    for (int ww = 0; ww < 8; ++ww) s += red[ww * 1024 + d];
    dst[d] = s;
  }
}

// ---------------------------------------------------------------------------
// Per-(b,head): reduce gram + sumsq partials, scale by temperature & inverse
// l2 norms, top-16 mask, softmax -> a1; then fold into the projection:
//   M_b[o, h*32+d] = attn1 * sum_c proj_w[o, h*32+c] * a1[c][d]   (bf16 out)
// ---------------------------------------------------------------------------
__global__ __launch_bounds__(256)
void attn_proj_kernel(const float* __restrict__ gramP,   // [b][h][32][1024]
                      const float* __restrict__ sumsqP,  // [b][384][32]
                      const float* __restrict__ temperature, // [6]
                      const float* __restrict__ attn1,   // [1]
                      const float* __restrict__ proj_w,  // [192,192] f32
                      unsigned short* __restrict__ Mout) // bf16 [b][192][192]
{
  __shared__ float sA[32][33];
  __shared__ float sP[32][33];
  __shared__ float rq[32], rk[32];
  const int t  = threadIdx.x;
  const int hd = blockIdx.x;
  const int b  = blockIdx.y;
  const float temp = temperature[hd];

  if (t < 64) {                                // inverse l2 norms
    const int sel = t >> 5, c = t & 31;
    const int chIdx = sel * 192 + hd * 32 + c;
    const float* p = sumsqP + (((long)b * 384 + chIdx) << 5);
    float s = 0.f;
    #pragma unroll
    for (int i = 0; i < 32; ++i) s += p[i];
    const float inv = 1.f / fmaxf(sqrtf(s), 1e-12f);
    if (sel == 0) rq[c] = inv; else rk[c] = inv;
  }
  __syncthreads();

  const float* gp = gramP + (((long)(b * HEADS + hd)) << 15);
  for (int d = t; d < 1024; d += 256) {        // reduce 32 slab partials
    float s = 0.f;
    #pragma unroll
    for (int sl = 0; sl < 32; ++sl) s += gp[(sl << 10) + d];
    const int c = d >> 5, j = d & 31;
    sA[c][j] = s * temp * rq[c] * rk[j];
  }
  __syncthreads();

  if (t < 32) {                                // per-row top-16 + softmax
    float v[32];
    for (int j = 0; j < 32; ++j) v[j] = sA[t][j];
    float kth = -3.4e38f;
    for (int j = 0; j < 32; ++j) {
      int gt = 0, ge = 0;
      for (int i = 0; i < 32; ++i) { gt += (v[i] > v[j]); ge += (v[i] >= v[j]); }
      if (gt <= 15 && ge >= 16) kth = v[j];    // 16th largest
    }
    float mx = -3.4e38f;
    for (int j = 0; j < 32; ++j) if (v[j] >= kth) mx = fmaxf(mx, v[j]);
    float e[32], sum = 0.f;
    for (int j = 0; j < 32; ++j) {
      const float ev = (v[j] >= kth) ? __expf(v[j] - mx) : 0.f;
      e[j] = ev; sum += ev;
    }
    const float inv = 1.f / sum;
    for (int j = 0; j < 32; ++j) sP[t][j] = e[j] * inv;
  }
  __syncthreads();

  const float a1s = attn1[0];
  for (int idx = t; idx < 192 * 32; idx += 256) {
    const int o = idx >> 5, d = idx & 31;
    float s = 0.f;
    #pragma unroll
    for (int c = 0; c < 32; ++c)
      s += proj_w[o * 192 + hd * 32 + c] * sP[c][d];
    s *= a1s;
    Mout[(long)b * 192 * 192 + (long)o * 192 + hd * 32 + d] =
        (unsigned short)f2bf1(s);
  }
}

// ---------------------------------------------------------------------------
// Host side
// ---------------------------------------------------------------------------
extern "C" void kernel_launch(void* const* d_in, const int* in_sizes, int n_in,
                              void* d_out, int out_size, void* d_ws, size_t ws_size,
                              hipStream_t stream) {
  const float* x      = (const float*)d_in[0]; // [2,192,256,256]
  const float* qkv_w  = (const float*)d_in[1]; // [576,192]
  const float* dw_w   = (const float*)d_in[2]; // [576,1,3,3]
  const float* proj_w = (const float*)d_in[3]; // [192,192]
  const float* temp   = (const float*)d_in[4]; // [6,1,1]
  const float* attn1  = (const float*)d_in[5]; // [1]
  float* out = (float*)d_out;                  // [2,192,256,256] f32

  // workspace layout (256B aligned)
  size_t off = 0;
  auto alloc = [&](size_t bytes) -> void* {
    void* p = (char*)d_ws + off;
    off += (bytes + 255) & ~(size_t)255;
    return p;
  };
  unsigned short* ws_qkv  = (unsigned short*)alloc((size_t)BATCH * MQKV * NPIX * 2);
  unsigned short* ws_qkvd = (unsigned short*)alloc((size_t)BATCH * MQKV * NPIX * 2);
  float*          ws_gram = (float*)alloc((size_t)BATCH * HEADS * 32 * 1024 * 4);
  float*          ws_ssq  = (float*)alloc((size_t)BATCH * 384 * 32 * 4);
  unsigned short* ws_M    = (unsigned short*)alloc((size_t)BATCH * 192 * 192 * 2);
  (void)ws_size; (void)in_sizes; (void)n_in; (void)out_size;

  // 1) qkv = qkv_w @ x   (A: f32 weights shared across batch; B: f32 x; out bf16)
  {
    dim3 grid(NPIX / 128, MQKV / 64, BATCH);
    gemm192_wmma<true, true, false, false><<<grid, 256, 0, stream>>>(
        qkv_w, x, ws_qkv,
        /*aStride=*/0,
        /*bStride=*/(size_t)CC * NPIX * sizeof(float),
        /*oStride=*/(size_t)MQKV * NPIX * sizeof(unsigned short));
  }

  // 2) depthwise 3x3 + k-transpose + sumsq partials
  {
    dim3 grid(32, MQKV, BATCH);
    dwconv_kernel<<<grid, 256, 0, stream>>>(ws_qkv, dw_w, ws_qkvd, ws_ssq);
  }

  // 3) Gram matrices q@kT (partials per n-slab), TDM-staged tiles
  {
    dim3 grid(32, HEADS, BATCH);
    gram_kernel<<<grid, 256, 0, stream>>>(ws_qkvd, ws_gram);
  }

  // 4) norms + top-16 softmax + fold (attn1, a1, proj_w) -> M_b (bf16 192x192)
  {
    dim3 grid(HEADS, BATCH);
    attn_proj_kernel<<<grid, 256, 0, stream>>>(ws_gram, ws_ssq, temp, attn1,
                                               proj_w, ws_M);
  }

  // 5) out = M_b @ v   (A: bf16 M_b per batch; B: bf16 v via TDM; out f32)
  {
    dim3 grid(NPIX / 128, CC / 64, BATCH);
    const unsigned short* vbase = ws_qkvd + (size_t)384 * NPIX; // batch 0 v
    gemm192_wmma<false, false, true, true><<<grid, 256, 0, stream>>>(
        ws_M, vbase, out,
        /*aStride=*/(size_t)192 * 192 * 2,
        /*bStride=*/(size_t)MQKV * NPIX * 2,
        /*oStride=*/(size_t)CC * NPIX * sizeof(float));
  }
}
